// DeepSeek3TransformerBlock_68332929679913
// MI455X (gfx1250) — matrix operand order repaired and checked by reading, not compiled, not run on hardware
//
#include <hip/hip_runtime.h>
#include <hip/hip_bf16.h>
#include <math.h>

// ---------------- problem constants ----------------
#define BB   2
#define SS   1024
#define HH   2048
#define NHH  16
#define DHH  128
#define RR   64
#define CQQ  1024
#define CKVV 512
#define EE   8
#define II   1024
#define TT   (BB * SS)
#define ATTN_SCALE 0.07216878364870322f   // 1/sqrt(DH + R) = 1/sqrt(192)

// ---------------- CDNA5 feature detection ----------------
#if defined(__has_builtin)
#  if __has_builtin(__builtin_amdgcn_tensor_load_to_lds)
#    define USE_TDM 1
#  endif
#endif
#ifndef USE_TDM
#  define USE_TDM 0
#endif
#if defined(__has_builtin)
#  if __has_builtin(__builtin_amdgcn_s_wait_tensorcnt)
#    define HAVE_WAIT_TENSOR 1
#  endif
#endif
#ifndef HAVE_WAIT_TENSOR
#  define HAVE_WAIT_TENSOR 0
#endif

// ---------------- WMMA helpers (CDNA5 gfx1250, wave32) ----------------
typedef __attribute__((ext_vector_type(16))) __bf16        v16bf;
typedef __attribute__((ext_vector_type(8)))  float         v8f;
typedef __attribute__((ext_vector_type(4)))  unsigned int  u32x4;
typedef __attribute__((ext_vector_type(8)))  int           i32x8;
typedef __attribute__((ext_vector_type(4)))  int           i32x4;

__device__ inline v8f wmma_bf16(v16bf a, v16bf b, v8f c) {
  // D = A(16x32 bf16) * B(32x16 bf16) + C(16x16 f32)
  return __builtin_amdgcn_wmma_f32_16x16x32_bf16(false, a, false, b, (short)0, c,
                                                 false, false);
}

// A-fragment: lane l holds row m=l&15; K(i) = i + (l>=16?16:0).
__device__ inline v16bf pack_a_frag(const float* __restrict__ p, int hs) {
  v16bf r;
#pragma unroll
  for (int i = 0; i < 16; ++i) r[i] = (__bf16)p[2 * hs + i];
  return r;
}

// B-fragment from 16 consecutive floats along K.
__device__ inline v16bf pack_b_frag16(const float* __restrict__ p) {
  v16bf r;
#pragma unroll
  for (int i = 0; i < 16; ++i) r[i] = (__bf16)p[i];
  return r;
}

// ---------------- Tensor Data Mover: 2-D tile global -> LDS ----------------
// D# per CDNA5 ISA ch.8: group0 = {count|lds_addr|global_addr|type=2},
// group1 = {data_size=4B, pad cfg, tensor/tile dims, dim0 stride}.
// pad_interval_code: 4 -> pad every 32 DWORDs, 5 -> every 64 DWORDs.
// pad_amount_code:   3 -> insert 4 DWORDs of padding.
__device__ inline void tdm_load_2d(unsigned lds_off_bytes, const void* gaddr,
                                   unsigned tile_w_elems, unsigned tile_h,
                                   unsigned row_stride_elems,
                                   unsigned pad_interval_code,
                                   unsigned pad_amount_code) {
#if USE_TDM
  unsigned long long ga  = (unsigned long long)(uintptr_t)gaddr;
  unsigned long long g0l = 1ull | ((unsigned long long)lds_off_bytes << 32);
  unsigned long long g0h = (ga & 0x01FFFFFFFFFFFFFFull) | (2ull << 62);
  u32x4 g0 = { (unsigned)g0l, (unsigned)(g0l >> 32),
               (unsigned)g0h, (unsigned)(g0h >> 32) };
  unsigned long long q0 = (2ull << 16)                     // data_size = 4B
                        | (1ull << 20)                     // pad_enable
                        | ((unsigned long long)pad_interval_code << 22)
                        | ((unsigned long long)pad_amount_code << 25)
                        | ((unsigned long long)(tile_w_elems & 0xFFFF) << 48);
  unsigned long long q1 = ((unsigned long long)(tile_w_elems >> 16) & 0xFFFF)
                        | ((unsigned long long)tile_h << 16)      // tensor_dim1
                        | ((unsigned long long)tile_w_elems << 48); // tile_dim0
  unsigned long long q2 = (unsigned long long)tile_h               // tile_dim1
                        | ((unsigned long long)row_stride_elems << 32);
  unsigned long long q3 = 0ull;                                    // stride hi, dim1_stride
  i32x8 g1 = { (int)q0, (int)(q0 >> 32), (int)q1, (int)(q1 >> 32),
               (int)q2, (int)(q2 >> 32), (int)q3, (int)(q3 >> 32) };
  i32x4 gz = { 0, 0, 0, 0 };
#  if __clang_major__ >= 23
  i32x8 gz8 = { 0, 0, 0, 0, 0, 0, 0, 0 };
  __builtin_amdgcn_tensor_load_to_lds(g0, g1, gz, gz, gz8, 0);
#  else
  __builtin_amdgcn_tensor_load_to_lds(g0, g1, gz, gz, 0);
#  endif
#else
  (void)lds_off_bytes; (void)gaddr; (void)tile_w_elems; (void)tile_h;
  (void)row_stride_elems; (void)pad_interval_code; (void)pad_amount_code;
#endif
}

// ---------------- RMSNorm (fp32, block per row) ----------------
__global__ __launch_bounds__(256) void rmsnorm_kernel(const float* __restrict__ in,
                                                      const float* __restrict__ w,
                                                      float* __restrict__ out, int W) {
  int row = blockIdx.x;
  const float* p = in + (size_t)row * W;
  float ss = 0.f;
  for (int j = threadIdx.x; j < W; j += blockDim.x) { float v = p[j]; ss += v * v; }
#pragma unroll
  for (int m = 16; m >= 1; m >>= 1) ss += __shfl_xor(ss, m, 32);
  __shared__ float red[8];
  int wid = threadIdx.x >> 5;
  if ((threadIdx.x & 31) == 0) red[wid] = ss;
  __syncthreads();
  float tot = 0.f;
  int nw = blockDim.x >> 5;
  for (int i = 0; i < nw; ++i) tot += red[i];
  float sc = rsqrtf(tot / (float)W + 1e-5f);
  float* o = out + (size_t)row * W;
  for (int j = threadIdx.x; j < W; j += blockDim.x) o[j] = p[j] * sc * w[j];
}

// ---------------- block-tiled WMMA GEMM with TDM-fed double-buffered LDS ----
// C[m][n] = scale_m * sum_k A[m][k]*B[k][n] (+ addsrc) (+ C if accum)
// Block 256 thr = 8 waves -> 128x64 C tile; wave -> 32x32 (2x2 WMMA frags).
// LDS tiles: A 128x32 f32 (pitch 36), B 32x64 f32 (pitch 68), double buffered.
#define BMT 128
#define BNT 64
#define BKT 32
#define AP  36   // padded A pitch (floats): 32 + 4   (TDM pad 32-DW interval)
#define BP  68   // padded B pitch (floats): 64 + 4   (TDM pad 64-DW interval)

__global__ __launch_bounds__(256) void gemm_wmma(const float* __restrict__ A,
                                                 const float* __restrict__ B,
                                                 float* __restrict__ C,
                                                 const float* __restrict__ addsrc,
                                                 const float* __restrict__ rowscale,
                                                 int rs_stride, float alpha,
                                                 int M, int N, int K, int accum) {
  __shared__ float As[2][BMT * AP];
  __shared__ float Bs[2][BKT * BP];

  int tid  = threadIdx.x;
  int lane = tid & 31;
  int wave = tid >> 5;        // 0..7
  int wm   = wave >> 1;       // 0..3  (M direction)
  int wn   = wave & 1;        // 0..1  (N direction)
  int m_blk = blockIdx.y * BMT;
  int n_blk = blockIdx.x * BNT;

  int nl    = lane & 15;
  int hsA   = (lane >> 4) << 3;   // 0 / 8
  int koffB = (lane >> 4) << 4;   // 0 / 16
  int rsel  = (lane >> 4) << 3;   // 0 / 8 (C row offset)

  v8f acc[2][2];
#pragma unroll
  for (int sm = 0; sm < 2; ++sm)
#pragma unroll
    for (int sn = 0; sn < 2; ++sn) acc[sm][sn] = (v8f){};

  const int nk = K / BKT;

  // ---- stage one K-tile into LDS buffer `buf` ----
  auto stage = [&](int buf, int kk) {
#if USE_TDM
    if (wave == 0) {
      tdm_load_2d((unsigned)(uintptr_t)&As[buf][0],
                  A + (size_t)m_blk * K + kk, BKT, BMT, (unsigned)K, 4u, 3u);
      tdm_load_2d((unsigned)(uintptr_t)&Bs[buf][0],
                  B + (size_t)kk * N + n_blk, BNT, BKT, (unsigned)N, 5u, 3u);
    }
#else
    // cooperative fallback: 256 threads, float4 copies
#pragma unroll
    for (int t = 0; t < 4; ++t) {               // A: 1024 float4
      int f = tid + t * 256;
      int row = f >> 3, c4 = f & 7;
      float4 d = *(const float4*)(A + (size_t)(m_blk + row) * K + kk + c4 * 4);
      *(float4*)&As[buf][row * AP + c4 * 4] = d;
    }
#pragma unroll
    for (int t = 0; t < 2; ++t) {               // B: 512 float4
      int f = tid + t * 256;
      int row = f >> 4, c4 = f & 15;
      float4 d = *(const float4*)(B + (size_t)(kk + row) * N + n_blk + c4 * 4);
      *(float4*)&Bs[buf][row * BP + c4 * 4] = d;
    }
#endif
  };

  stage(0, 0);
  for (int i = 0; i < nk; ++i) {
    int buf = i & 1;
#if USE_TDM && HAVE_WAIT_TENSOR
    if (wave == 0) __builtin_amdgcn_s_wait_tensorcnt(0);
#endif
    __syncthreads();                     // tile i visible to all waves
    if (i + 1 < nk) stage(buf ^ 1, (i + 1) * BKT);

    const float* as = &As[buf][0];
    const float* bs = &Bs[buf][0];
    v16bf af[2], bfr[2];
#pragma unroll
    for (int sm = 0; sm < 2; ++sm)
      af[sm] = pack_a_frag(as + (wm * 32 + sm * 16 + nl) * AP, hsA);
#pragma unroll
    for (int sn = 0; sn < 2; ++sn) {
      const float* bp = bs + wn * 32 + sn * 16 + nl;
      v16bf r;
#pragma unroll
      for (int e = 0; e < 16; ++e) r[e] = (__bf16)bp[(koffB + e) * BP];
      bfr[sn] = r;
    }
#pragma unroll
    for (int sm = 0; sm < 2; ++sm)
#pragma unroll
      for (int sn = 0; sn < 2; ++sn)
        acc[sm][sn] = wmma_bf16(af[sm], bfr[sn], acc[sm][sn]);
    __syncthreads();                     // done reading before buf is reused
  }

  // ---- epilogue ----
#pragma unroll
  for (int sm = 0; sm < 2; ++sm) {
#pragma unroll
    for (int v = 0; v < 8; ++v) {
      int m = m_blk + wm * 32 + sm * 16 + rsel + v;
      float sc = rowscale ? rowscale[(size_t)m * rs_stride] : alpha;
      size_t rowoff = (size_t)m * N;
#pragma unroll
      for (int sn = 0; sn < 2; ++sn) {
        int n = n_blk + wn * 32 + sn * 16 + nl;
        size_t off = rowoff + n;
        float val = acc[sm][sn][v] * sc;
        if (addsrc) val += addsrc[off];
        if (accum)  val += C[off];
        C[off] = val;
      }
    }
  }
}

// ---------------- RoPE (rotate-half, in place) ----------------
__global__ __launch_bounds__(256) void rope_kernel(float* __restrict__ buf, int nh,
                                                   int total) {
  int idx = blockIdx.x * blockDim.x + threadIdx.x;
  if (idx >= total) return;
  const int half = RR / 2;
  int i    = idx % half;
  int rest = idx / half;
  int hh   = rest % nh;
  int t    = rest / nh;
  int pos  = t % SS;
  float freq = powf(10000.f, -(float)(2 * i) / (float)RR);
  float ang  = (float)pos * freq;
  float s, c;
  sincosf(ang, &s, &c);
  float* p = buf + ((size_t)t * nh + hh) * RR;
  float x1 = p[i], x2 = p[i + half];
  p[i]        = x1 * c - x2 * s;
  p[i + half] = x2 * c + x1 * s;
}

// ---------------- MLA flash attention (1 wave / (b,h,16 queries)) ----------------
__global__ __launch_bounds__(32) void mla_attn_kernel(const float* __restrict__ q,
                                                      const float* __restrict__ qr,
                                                      const float* __restrict__ k,
                                                      const float* __restrict__ v,
                                                      const float* __restrict__ kr,
                                                      float* __restrict__ out) {
  int lane = threadIdx.x;
  int q0   = blockIdx.x * 16;
  int h    = blockIdx.y;
  int b    = blockIdx.z;
  int tb   = b * SS;
  int nl    = lane & 15;
  int hsA   = (lane >> 4) << 3;
  int koffB = (lane >> 4) << 4;
  int rsel  = (lane >> 4) << 3;

  const float* qrow  = q  + ((size_t)(tb + q0 + nl) * NHH + h) * DHH;
  const float* qrrow = qr + ((size_t)(tb + q0 + nl) * NHH + h) * RR;
  v16bf aq[4], aqr[2];
#pragma unroll
  for (int t = 0; t < 4; ++t) aq[t]  = pack_a_frag(qrow  + 32 * t, hsA);
#pragma unroll
  for (int t = 0; t < 2; ++t) aqr[t] = pack_a_frag(qrrow + 32 * t, hsA);

  v8f o[8];
#pragma unroll
  for (int t = 0; t < 8; ++t) o[t] = (v8f){};
  float mprev[8], lsum[8];
#pragma unroll
  for (int t = 0; t < 8; ++t) { mprev[t] = -3.0e38f; lsum[t] = 0.f; }

  __shared__ __bf16 Pt[16][32];

  int jmax = (q0 + 15) / 32;
  for (int j = 0; j <= jmax; ++j) {
    int kb = j * 32;
    int key0 = tb + kb + nl;
    const float* kp0  = k  + ((size_t)key0        * NHH + h) * DHH;
    const float* kp1  = k  + ((size_t)(key0 + 16) * NHH + h) * DHH;
    const float* krp0 = kr + (size_t)key0        * RR;
    const float* krp1 = kr + (size_t)(key0 + 16) * RR;

    v8f s0 = {}, s1 = {};
#pragma unroll
    for (int t = 0; t < 4; ++t) {
      s0 = wmma_bf16(aq[t], pack_b_frag16(kp0 + 32 * t + koffB), s0);
      s1 = wmma_bf16(aq[t], pack_b_frag16(kp1 + 32 * t + koffB), s1);
    }
#pragma unroll
    for (int t = 0; t < 2; ++t) {
      s0 = wmma_bf16(aqr[t], pack_b_frag16(krp0 + 32 * t + koffB), s0);
      s1 = wmma_bf16(aqr[t], pack_b_frag16(krp1 + 32 * t + koffB), s1);
    }

    int col0 = kb + nl, col1 = kb + 16 + nl;
    float p0[8], p1[8], alp[8];
#pragma unroll
    for (int vv = 0; vv < 8; ++vv) {
      int row = q0 + rsel + vv;
      float x0 = s0[vv] * ATTN_SCALE; if (col0 > row) x0 = -3.0e38f;
      float x1 = s1[vv] * ATTN_SCALE; if (col1 > row) x1 = -3.0e38f;
      float rm = fmaxf(x0, x1);
#pragma unroll
      for (int msk = 8; msk >= 1; msk >>= 1) rm = fmaxf(rm, __shfl_xor(rm, msk, 32));
      float mn = fmaxf(mprev[vv], rm);
      float al = __expf(mprev[vv] - mn);
      float e0 = __expf(x0 - mn);
      float e1 = __expf(x1 - mn);
      float rs = e0 + e1;
#pragma unroll
      for (int msk = 8; msk >= 1; msk >>= 1) rs += __shfl_xor(rs, msk, 32);
      lsum[vv]  = lsum[vv] * al + rs;
      mprev[vv] = mn;
      p0[vv] = e0; p1[vv] = e1; alp[vv] = al;
    }
#pragma unroll
    for (int t = 0; t < 8; ++t)
#pragma unroll
      for (int vv = 0; vv < 8; ++vv) o[t][vv] *= alp[vv];

    // C-layout -> A-layout for P via LDS (16x32 bf16)
#pragma unroll
    for (int vv = 0; vv < 8; ++vv) {
      Pt[rsel + vv][nl]      = (__bf16)p0[vv];
      Pt[rsel + vv][16 + nl] = (__bf16)p1[vv];
    }
    __syncthreads();
    v16bf pa;
#pragma unroll
    for (int i = 0; i < 16; ++i) pa[i] = Pt[nl][2 * hsA + i];
    __syncthreads();

    const float* vbase = v + ((size_t)(tb + kb + koffB) * NHH + h) * DHH + nl;
#pragma unroll
    for (int t = 0; t < 8; ++t) {
      v16bf bv;
#pragma unroll
      for (int i = 0; i < 16; ++i)
        bv[i] = (__bf16)vbase[(size_t)i * (NHH * DHH) + 16 * t];
      o[t] = wmma_bf16(pa, bv, o[t]);
    }
  }

#pragma unroll
  for (int vv = 0; vv < 8; ++vv) {
    float inv = 1.0f / lsum[vv];
    int row = q0 + rsel + vv;
    float* op = out + ((size_t)(tb + row) * NHH + h) * DHH + nl;
#pragma unroll
    for (int t = 0; t < 8; ++t) op[16 * t] = o[t][vv] * inv;
  }
}

// ---------------- MoE gate: sigmoid + biased top-2 (1 wave / token) ----------------
__global__ __launch_bounds__(32) void gate_kernel(const float* __restrict__ h2,
                                                  const float* __restrict__ cent,
                                                  const float* __restrict__ bias,
                                                  float* __restrict__ dw) {
  int t = blockIdx.x;
  int lane = threadIdx.x;
  const float* hp = h2 + (size_t)t * HH;
  float sc[EE];
#pragma unroll
  for (int e = 0; e < EE; ++e) {
    const float* cp = cent + (size_t)e * HH;
    float s = 0.f;
    for (int j = lane; j < HH; j += 32) s += hp[j] * cp[j];
#pragma unroll
    for (int m = 16; m >= 1; m >>= 1) s += __shfl_xor(s, m, 32);
    sc[e] = 1.f / (1.f + __expf(-s));
  }
  if (lane == 0) {
    int e1 = 0; float b1 = sc[0] + bias[0];
    for (int e = 1; e < EE; ++e) { float v = sc[e] + bias[e]; if (v > b1) { b1 = v; e1 = e; } }
    int e2 = -1; float b2 = -3.0e38f;
    for (int e = 0; e < EE; ++e) {
      if (e == e1) continue;
      float v = sc[e] + bias[e];
      if (v > b2) { b2 = v; e2 = e; }
    }
    float g1 = sc[e1], g2 = sc[e2], den = g1 + g2;
    for (int e = 0; e < EE; ++e) {
      float wv = (e == e1) ? g1 / den : (e == e2) ? g2 / den : 0.f;
      dw[(size_t)t * EE + e] = wv;
    }
  }
}

// ---------------- SiLU(g1) * g3 ----------------
__global__ __launch_bounds__(256) void silu_mul_kernel(const float* __restrict__ g1,
                                                       const float* __restrict__ g3,
                                                       float* __restrict__ act, int n) {
  int i = blockIdx.x * blockDim.x + threadIdx.x;
  if (i < n) {
    float a = g1[i];
    act[i] = (a / (1.f + __expf(-a))) * g3[i];
  }
}

// ---------------- orchestration ----------------
static inline void launch_gemm(const float* A, const float* B, float* C,
                               const float* addsrc, const float* rowscale,
                               int rs_stride, float alpha, int M, int N, int K,
                               int accum, hipStream_t stream) {
  dim3 grid(N / BNT, M / BMT);
  hipLaunchKernelGGL(gemm_wmma, grid, dim3(256), 0, stream,
                     A, B, C, addsrc, rowscale, rs_stride, alpha, M, N, K, accum);
}

extern "C" void kernel_launch(void* const* d_in, const int* in_sizes, int n_in,
                              void* d_out, int out_size, void* d_ws, size_t ws_size,
                              hipStream_t stream) {
  const float* x           = (const float*)d_in[0];
  const float* attn_norm_w = (const float*)d_in[1];
  const float* ffn_norm_w  = (const float*)d_in[2];
  const float* W_dq        = (const float*)d_in[3];
  const float* q_norm_w    = (const float*)d_in[4];
  const float* W_uq        = (const float*)d_in[5];
  const float* W_qr        = (const float*)d_in[6];
  const float* W_dkv       = (const float*)d_in[7];
  const float* kv_norm_w   = (const float*)d_in[8];
  const float* W_uk        = (const float*)d_in[9];
  const float* W_uv        = (const float*)d_in[10];
  const float* W_kr        = (const float*)d_in[11];
  const float* W_o         = (const float*)d_in[12];
  const float* centroids   = (const float*)d_in[13];
  const float* expert_bias = (const float*)d_in[14];
  const float* w1          = (const float*)d_in[15];
  const float* w3          = (const float*)d_in[16];
  const float* w2          = (const float*)d_in[17];
  const float* w1s         = (const float*)d_in[18];
  const float* w3s         = (const float*)d_in[19];
  const float* w2s         = (const float*)d_in[20];
  float* out = (float*)d_out;
  float* ws  = (float*)d_ws;

  const size_t M1 = 1024 * 1024;
  float* hn    = ws;                         // 4M floats  (later h2)
  float* qb    = ws + 4 * M1;                // 4M
  float* kb    = ws + 8 * M1;                // 4M
  float* vb    = ws + 12 * M1;               // 4M
  float* qrb   = ws + 16 * M1;               // 2M (later g1)
  float* cqb   = ws + 18 * M1;               // 2M (later g3)
  float* ckvb  = ws + 20 * M1;               // 2M region (ckv, later act)
  float* krb   = ws + 22 * M1;               // 256K region (kr, later dense_w)
  float* attnb = ws + 22 * M1 + 262144;      // 4M

  // ---- MLA attention ----
  hipLaunchKernelGGL(rmsnorm_kernel, dim3(TT), dim3(256), 0, stream,
                     x, attn_norm_w, hn, HH);
  launch_gemm(hn, W_dq, cqb, nullptr, nullptr, 1, 1.f, TT, CQQ, HH, 0, stream);
  hipLaunchKernelGGL(rmsnorm_kernel, dim3(TT), dim3(256), 0, stream,
                     cqb, q_norm_w, cqb, CQQ);
  launch_gemm(cqb, W_uq, qb,  nullptr, nullptr, 1, 1.f, TT, NHH * DHH, CQQ, 0, stream);
  launch_gemm(cqb, W_qr, qrb, nullptr, nullptr, 1, 1.f, TT, NHH * RR,  CQQ, 0, stream);
  launch_gemm(hn, W_dkv, ckvb, nullptr, nullptr, 1, 1.f, TT, CKVV, HH, 0, stream);
  hipLaunchKernelGGL(rmsnorm_kernel, dim3(TT), dim3(256), 0, stream,
                     ckvb, kv_norm_w, ckvb, CKVV);
  launch_gemm(ckvb, W_uk, kb, nullptr, nullptr, 1, 1.f, TT, NHH * DHH, CKVV, 0, stream);
  launch_gemm(ckvb, W_uv, vb, nullptr, nullptr, 1, 1.f, TT, NHH * DHH, CKVV, 0, stream);
  launch_gemm(hn, W_kr, krb, nullptr, nullptr, 1, 1.f, TT, RR, HH, 0, stream);

  int ntot_q = TT * NHH * (RR / 2);
  hipLaunchKernelGGL(rope_kernel, dim3((ntot_q + 255) / 256), dim3(256), 0, stream,
                     qrb, NHH, ntot_q);
  int ntot_k = TT * (RR / 2);
  hipLaunchKernelGGL(rope_kernel, dim3((ntot_k + 255) / 256), dim3(256), 0, stream,
                     krb, 1, ntot_k);

  hipLaunchKernelGGL(mla_attn_kernel, dim3(SS / 16, NHH, BB), dim3(32), 0, stream,
                     qb, qrb, kb, vb, krb, attnb);

  // out = x + attn @ W_o   (residual fused into GEMM epilogue)
  launch_gemm(attnb, W_o, out, x, nullptr, 1, 1.f, TT, HH, NHH * DHH, 0, stream);

  // ---- MoE ----
  float* h2      = hn;    // reuse
  float* g1buf   = qrb;   // reuse
  float* g3buf   = cqb;   // reuse
  float* actbuf  = ckvb;  // reuse
  float* dense_w = krb;   // reuse

  hipLaunchKernelGGL(rmsnorm_kernel, dim3(TT), dim3(256), 0, stream,
                     out, ffn_norm_w, h2, HH);
  hipLaunchKernelGGL(gate_kernel, dim3(TT), dim3(32), 0, stream,
                     h2, centroids, expert_bias, dense_w);

  const int nact = TT * II;
  for (int e = 0; e < EE; ++e) {
    const float* w1e = w1 + (size_t)e * HH * II;
    const float* w3e = w3 + (size_t)e * HH * II;
    const float* w2e = w2 + (size_t)e * II * HH;
    launch_gemm(h2, w1e, g1buf, nullptr, nullptr, 1, 1.f, TT, II, HH, 0, stream);
    launch_gemm(h2, w3e, g3buf, nullptr, nullptr, 1, 1.f, TT, II, HH, 0, stream);
    hipLaunchKernelGGL(silu_mul_kernel, dim3((nact + 255) / 256), dim3(256), 0, stream,
                       g1buf, g3buf, actbuf, nact);
    launch_gemm(actbuf, w2e, out, nullptr, dense_w + e, EE, 1.f, TT, HH, II, 1, stream);
  }
  for (int si = 0; si < 2; ++si) {
    const float* w1e = w1s + (size_t)si * HH * II;
    const float* w3e = w3s + (size_t)si * HH * II;
    const float* w2e = w2s + (size_t)si * II * HH;
    launch_gemm(h2, w1e, g1buf, nullptr, nullptr, 1, 1.f, TT, II, HH, 0, stream);
    launch_gemm(h2, w3e, g3buf, nullptr, nullptr, 1, 1.f, TT, II, HH, 0, stream);
    hipLaunchKernelGGL(silu_mul_kernel, dim3((nact + 255) / 256), dim3(256), 0, stream,
                       g1buf, g3buf, actbuf, nact);
    launch_gemm(actbuf, w2e, out, nullptr, nullptr, 1, 0.1f, TT, HH, II, 1, stream);
  }
  (void)in_sizes; (void)n_in; (void)out_size; (void)ws_size;
}
